// LidarToBEV_80083960201741
// MI455X (gfx1250) — matrix operand order repaired and checked by reading, not compiled, not run on hardware
//
#include <hip/hip_runtime.h>
#include <hip/hip_bf16.h>

// ---------------------------------------------------------------------------
// LidarToBEV fused pipeline for MI455X (gfx1250, wave32, WMMA + TDM)
//   k1: zero BEV grid
//   k2: scatter-max rasterize (atomicMax on float bits, values >= 0)
//   k3: log1p + zero-pad to f16 [B][8][402][402]
//   k4/k5: repack conv weights into WMMA-B-friendly K-padded layouts (f16)
//   k6: fused im2col-GEMM (3x3, K=72->96) + bias + ReLU (in LDS, f16)
//       -> GEMM (1x1, K=128) + bias -> transposed coalesced f32 stores
//       Weights staged to LDS via TENSOR_LOAD_TO_LDS (TDM), overlapped with
//       the VALU im2col gather; synced with s_wait_tensorcnt + barrier.
// ---------------------------------------------------------------------------

typedef __attribute__((ext_vector_type(16))) _Float16 v16h;
typedef __attribute__((ext_vector_type(8)))  _Float16 v8h;
typedef __attribute__((ext_vector_type(8)))  float    v8f;
typedef __attribute__((ext_vector_type(4)))  unsigned int u32x4;
typedef __attribute__((ext_vector_type(8)))  int      i32x8;
typedef __attribute__((ext_vector_type(4)))  int      i32x4;

#define BATCH 4
#define NPTS  200000
#define HW    400
#define HP    402            // padded spatial (1-px halo)
#define CIN   8              // 4 height bins * 2 raw feats
#define C1    128
#define C2    256
#define AS_STR 104           // K-stride stage-A operands (K=72 pad 96, +bank pad)
#define HS_STR 136           // K-stride stage-B operands (K=128, +bank pad)
#define OS_STR 20            // float stride of per-wave output staging rows

// workspace layout (bytes); total ~30.9 MB
#define SZ_GRID  ((size_t)BATCH*CIN*HW*HW*4)        // 20,480,000
#define OFF_BEVP (SZ_GRID)
#define SZ_BEVP  ((size_t)BATCH*CIN*HP*HP*2)        // 10,342,656
#define OFF_WB1  (OFF_BEVP + SZ_BEVP)
#define SZ_WB1   ((size_t)C1*AS_STR*2)              // 26,624 (8B aligned)
#define OFF_WB2  (OFF_WB1 + SZ_WB1)
#define SZ_WB2   ((size_t)C2*HS_STR*2)              // 69,632 (8B aligned)

// dynamic LDS layout (bytes)
#define SM_AS   0
#define SM_B1S  (64*AS_STR*2)                       // 13,312
#define SM_B2S  (SM_B1S + C1*AS_STR*2)              // 39,936
#define SM_HS   (SM_B2S + C2*HS_STR*2)              // 109,568
#define SM_OS   (SM_HS + 64*HS_STR*2)               // 126,976
#define SMEM_BYTES (SM_OS + 8*16*OS_STR*4)          // 137,216

__global__ void k_fill_zero(float* __restrict__ p, int n) {
    int i = blockIdx.x * blockDim.x + threadIdx.x;
    if (i < n) p[i] = 0.0f;
}

__global__ void k_rasterize(const float* __restrict__ pts, float* __restrict__ grid) {
    int idx = blockIdx.x * blockDim.x + threadIdx.x;
    if (idx >= BATCH * NPTS) return;
    int b = idx / NPTS;
    const float* p = pts + (size_t)idx * 4;
    float x = p[0], y = p[1], z = p[2], it = p[3];
    if (x < -50.0f || x > 50.0f || y < -50.0f || y > 50.0f || z < 0.0f || z > 5.0f) return;
    int xi = (int)((x + 50.0f) * 4.0f); xi = xi < 0 ? 0 : (xi > HW - 1 ? HW - 1 : xi);
    int yi = (int)((y + 50.0f) * 4.0f); yi = yi < 0 ? 0 : (yi > HW - 1 ? HW - 1 : yi);
    // searchsorted(side='left') == count of boundaries strictly < z
    int hb = (z > 1.25f) + (z > 2.5f) + (z > 3.75f);
    unsigned* g = (unsigned*)grid;
    int base = ((b * CIN + hb * 2) * HW + yi) * HW + xi;
    atomicMax(&g[base], __float_as_uint(1.0f));            // density
    atomicMax(&g[base + HW * HW], __float_as_uint(it));    // intensity
}

__global__ void k_log1p_pad(const float* __restrict__ grid, _Float16* __restrict__ bevp) {
    int idx = blockIdx.x * blockDim.x + threadIdx.x;
    if (idx >= BATCH * CIN * HP * HP) return;
    int ix = idx % HP;
    int r  = idx / HP;
    int iy = r % HP;
    int bc = r / HP;
    float v = 0.0f;
    if (iy >= 1 && iy <= HW && ix >= 1 && ix <= HW)
        v = log1pf(grid[(bc * HW + (iy - 1)) * HW + (ix - 1)]);
    bevp[idx] = (_Float16)v;
}

// w1 [128][8][3][3] -> wB1 [co][k] f16, k = ci*9 + ky*3 + kx, zero-padded to AS_STR
__global__ void k_prep_w1(const float* __restrict__ w1, _Float16* __restrict__ wB1) {
    int idx = blockIdx.x * blockDim.x + threadIdx.x;
    if (idx >= C1 * AS_STR) return;
    int co = idx / AS_STR, k = idx % AS_STR;
    _Float16 v = (_Float16)0.0f;
    if (k < CIN * 9) {
        int ci = k / 9, t = k - ci * 9;
        v = (_Float16)w1[(co * CIN + ci) * 9 + t];
    }
    wB1[idx] = v;
}

// w2 [256][128][1][1] -> wB2 [c2][k] f16, zero-padded to HS_STR
__global__ void k_prep_w2(const float* __restrict__ w2, _Float16* __restrict__ wB2) {
    int idx = blockIdx.x * blockDim.x + threadIdx.x;
    if (idx >= C2 * HS_STR) return;
    int c2 = idx / HS_STR, k = idx % HS_STR;
    _Float16 v = (_Float16)0.0f;
    if (k < C1) v = (_Float16)w2[c2 * C1 + k];
    wB2[idx] = v;
}

union Frag { v16h v; v8h h[2]; };

// Issue one TDM load: contiguous n8 x 8-byte elements, global -> LDS.
// D# per CDNA5 ISA ch.8: group0 {count=1, lds_addr, global_addr, type=2},
// group1 {data_size=3(8B), tensor_dim0=n8, tensor_dim1=1, tile_dim0=n8,
//         tile_dim1=1, dim0_stride=n8}; groups 2/3 zero (<=2D tensor).
__device__ static inline void tdm_copy_to_lds(unsigned lds_off, const void* gptr,
                                              unsigned n8) {
    unsigned long long ga = (unsigned long long)(size_t)gptr;
    u32x4 g0;
    g0[0] = 1u;                                            // count=1, user mode
    g0[1] = lds_off;                                       // lds_addr
    g0[2] = (unsigned)(ga & 0xffffffffu);                  // global_addr[31:0]
    g0[3] = (unsigned)((ga >> 32) & 0x01ffffffu) | (2u << 30); // ga[56:32] | type=2
    i32x8 g1;
    g1[0] = (int)(3u << 16);                               // mask=0, data_size=8B
    g1[1] = (int)((n8 & 0xffffu) << 16);                   // tensor_dim0[15:0]
    g1[2] = (int)(((n8 >> 16) & 0xffffu) | (1u << 16));    // dim0[31:16], tensor_dim1=1
    g1[3] = (int)((n8 & 0xffffu) << 16);                   // tile_dim0 = n8
    g1[4] = 1;                                             // tile_dim1=1, tile_dim2=0
    g1[5] = (int)n8;                                       // tensor_dim0_stride lo
    g1[6] = 0;
    g1[7] = 0;
    i32x4 z4 = {0, 0, 0, 0};
#if defined(__clang_major__) && (__clang_major__ >= 23)
    i32x8 z8 = {0, 0, 0, 0, 0, 0, 0, 0};
    __builtin_amdgcn_tensor_load_to_lds(g0, g1, z4, z4, z8, 0);
#else
    __builtin_amdgcn_tensor_load_to_lds(g0, g1, z4, z4, 0);
#endif
}

__global__ __launch_bounds__(256)
void k_fused_conv(const _Float16* __restrict__ bevp,
                  const _Float16* __restrict__ wB1,
                  const _Float16* __restrict__ wB2,
                  const float* __restrict__ b1,
                  const float* __restrict__ b2,
                  float* __restrict__ out) {
    extern __shared__ char smem[];
    _Float16* As  = (_Float16*)(smem + SM_AS);   // [64 px][AS_STR]  im2col A
    _Float16* B1s = (_Float16*)(smem + SM_B1S);  // [128 co][AS_STR] conv1 B
    _Float16* B2s = (_Float16*)(smem + SM_B2S);  // [256 c2][HS_STR] conv2 B
    _Float16* Hs  = (_Float16*)(smem + SM_HS);   // [64 px][HS_STR]  relu(conv1)
    float*    Os  = (float*)  (smem + SM_OS);    // [8 waves][16][OS_STR]

    const int tid   = threadIdx.x;
    const int lane  = tid & 31;
    const int wave  = tid >> 5;
    const int xbase = blockIdx.x * 16;           // 25 tiles of 16
    const int ybase = blockIdx.y * 4;            // 100 groups of 4 rows
    const int b     = blockIdx.z;

    // ---- TDM: stage both weight matrices into LDS (wave 0 issues; TDM
    //      ignores EXEC and runs async while all waves do the im2col) ----
    if (wave == 0) {
        tdm_copy_to_lds((unsigned)(size_t)(smem + SM_B1S), wB1, (unsigned)(SZ_WB1 >> 3));
        tdm_copy_to_lds((unsigned)(size_t)(smem + SM_B2S), wB2, (unsigned)(SZ_WB2 >> 3));
    }

    // ---- im2col gather: 64 pixels x 96 K (K 72..95 zero) ----
    for (int e = tid; e < 96 * 64; e += 256) {
        int k = e >> 6, p = e & 63;              // p fast -> coalesced global reads
        _Float16 v = (_Float16)0.0f;
        if (k < CIN * 9) {
            int ci = k / 9, t = k - ci * 9;
            int ky = t / 3, kx = t - ky * 3;
            int yy = ybase + (p >> 4) + ky;      // padded coords
            int xx = xbase + (p & 15) + kx;
            v = bevp[((b * CIN + ci) * HP + yy) * HP + xx];
        }
        As[p * AS_STR + k] = v;
    }
    if (wave == 0) __builtin_amdgcn_s_wait_tensorcnt(0);
    __syncthreads();

    const int row = lane & 15;                   // A row / B col / D col
    const int hiA = (lane >> 4) << 3;            // A K-base offset (0/8) == D M-offset
    const int hiB = (lane >> 4) << 4;            // unused; B frags also use hiA (0/8) as K-base
    const int mt  = wave >> 1;                   // wave-invariant M-tile (both stages)

    // ---- stage A: conv1 as GEMM, M=64 px, N=128, K=96 (3 k-steps) ----
    {
        Frag a[3];
        const _Float16* ap = As + (mt * 16 + row) * AS_STR;
#pragma unroll
        for (int ks = 0; ks < 3; ++ks) {         // A frags loaded once per wave
            a[ks].h[0] = *(const v8h*)(ap + hiA + ks * 32);
            a[ks].h[1] = *(const v8h*)(ap + hiA + ks * 32 + 16);
        }
        const int ntbase = (wave & 1) * 4;
        for (int i = 0; i < 4; ++i) {
            int nt = ntbase + i;
            v8f acc; for (int r = 0; r < 8; ++r) acc[r] = 0.0f;
            const _Float16* bp = B1s + (nt * 16 + row) * AS_STR;
            Frag bb[3];
#pragma unroll
            for (int ks = 0; ks < 3; ++ks) {
                bb[ks].h[0] = *(const v8h*)(bp + hiA + ks * 32);
                bb[ks].h[1] = *(const v8h*)(bp + hiA + ks * 32 + 16);
            }
#pragma unroll
            for (int ks = 0; ks < 3; ++ks)
                acc = __builtin_amdgcn_wmma_f32_16x16x32_f16(
                    false, a[ks].v, false, bb[ks].v, (short)0, acc, false, false);
            int co = nt * 16 + row;
            float bias = b1[co];
#pragma unroll
            for (int r = 0; r < 8; ++r) {
                float h = acc[r] + bias;
                h = h > 0.0f ? h : 0.0f;
                Hs[(mt * 16 + r + hiA) * HS_STR + co] = (_Float16)h;
            }
        }
    }
    __syncthreads();

    // ---- stage B: conv2 1x1 as GEMM, M=64 px, N=256, K=128 (4 k-steps) ----
    {
        Frag a[4];
        const _Float16* ap = Hs + (mt * 16 + row) * HS_STR;
#pragma unroll
        for (int ks = 0; ks < 4; ++ks) {         // A frags loaded once per wave
            a[ks].h[0] = *(const v8h*)(ap + hiA + ks * 32);
            a[ks].h[1] = *(const v8h*)(ap + hiA + ks * 32 + 16);
        }
        float* myOs = Os + wave * (16 * OS_STR);
        const int ntbase = (wave & 1) * 8;
        for (int i = 0; i < 8; ++i) {
            int nt = ntbase + i;
            v8f acc; for (int r = 0; r < 8; ++r) acc[r] = 0.0f;
            const _Float16* bp = B2s + (nt * 16 + row) * HS_STR;
            Frag bb[4];
#pragma unroll
            for (int ks = 0; ks < 4; ++ks) {
                bb[ks].h[0] = *(const v8h*)(bp + hiA + ks * 32);
                bb[ks].h[1] = *(const v8h*)(bp + hiA + ks * 32 + 16);
            }
#pragma unroll
            for (int ks = 0; ks < 4; ++ks)
                acc = __builtin_amdgcn_wmma_f32_16x16x32_f16(
                    false, a[ks].v, false, bb[ks].v, (short)0, acc, false, false);
            int c2 = nt * 16 + row;
            float bias = b2[c2];
            // stash D tile (per-wave region): Os[c2_local][x_local]
#pragma unroll
            for (int r = 0; r < 8; ++r)
                myOs[row * OS_STR + r + hiA] = acc[r] + bias;
            // cross-lane LDS RAW within the wave: DS ops are in-order per
            // wave; wait + compiler memory barrier before transposed readback
            asm volatile("s_wait_dscnt 0" ::: "memory");
            int c2l = lane >> 1;
            int xh  = (lane & 1) << 3;
            const float* src = myOs + c2l * OS_STR + xh;   // 16B aligned (OS_STR=20)
            float4 v0 = *(const float4*)(src);
            float4 v1 = *(const float4*)(src + 4);
            int c2g = nt * 16 + c2l;
            int yg  = ybase + mt;
            float* op = out + (((size_t)b * C2 + c2g) * HW + yg) * HW + xbase + xh;
            *(float4*)(op)     = v0;                       // 32B contiguous per lane
            *(float4*)(op + 4) = v1;
        }
    }
}

extern "C" void kernel_launch(void* const* d_in, const int* in_sizes, int n_in,
                              void* d_out, int out_size, void* d_ws, size_t ws_size,
                              hipStream_t stream) {
    (void)in_sizes; (void)n_in; (void)out_size; (void)ws_size;
    const float* pts = (const float*)d_in[0];
    const float* w1  = (const float*)d_in[1];
    const float* b1  = (const float*)d_in[2];
    const float* w2  = (const float*)d_in[3];
    const float* b2  = (const float*)d_in[4];
    float* out = (float*)d_out;

    char* ws = (char*)d_ws;                    // needs ~30.9 MB
    float*    grid = (float*)ws;
    _Float16* bevp = (_Float16*)(ws + OFF_BEVP);
    _Float16* wB1  = (_Float16*)(ws + OFF_WB1);
    _Float16* wB2  = (_Float16*)(ws + OFF_WB2);

    const int ngrid = BATCH * CIN * HW * HW;   // 5,120,000
    k_fill_zero<<<(ngrid + 255) / 256, 256, 0, stream>>>(grid, ngrid);
    k_rasterize<<<(BATCH * NPTS + 255) / 256, 256, 0, stream>>>(pts, grid);
    const int npad = BATCH * CIN * HP * HP;
    k_log1p_pad<<<(npad + 255) / 256, 256, 0, stream>>>(grid, bevp);
    k_prep_w1<<<(C1 * AS_STR + 255) / 256, 256, 0, stream>>>(w1, wB1);
    k_prep_w2<<<(C2 * HS_STR + 255) / 256, 256, 0, stream>>>(w2, wB2);

    hipFuncSetAttribute((const void*)k_fused_conv,
                        hipFuncAttributeMaxDynamicSharedMemorySize, SMEM_BYTES);
    dim3 grd(HW / 16, HW / 4, BATCH);          // (25, 100, 4)
    k_fused_conv<<<grd, 256, SMEM_BYTES, stream>>>(bevp, wB1, wB2, b1, b2, out);
}
